// EQGATEdgeGNN_30356828848149
// MI455X (gfx1250) — matrix operand order repaired and checked by reading, not compiled, not run on hardware
//
#include <hip/hip_runtime.h>
#include <math.h>

// ---------------- problem constants (from reference) ----------------
#define NB_   16            // batches
#define NPG_  48            // nodes per group
#define NN_   768           // N = NB*NPG
#define NE_   36096         // E = NB*NPG*(NPG-1)
#define K_    4             // knn K
#define EK_   3072          // NN_*K_
#define SI_   256
#define VI_   64
#define ED_   64
#define W1K_  580           // 2*SI + ED + 4
#define W1KP_ 608           // W1K padded to multiple of 32 (f16 WMMA K-chunk)
#define W2N_  513           // 3*VI + SI + 1 + ED

typedef __attribute__((ext_vector_type(2)))  float    v2f;
typedef __attribute__((ext_vector_type(8)))  float    v8f;
typedef _Float16 half_t;
typedef __attribute__((ext_vector_type(16))) _Float16 v16h;

// fp32 WMMA 16x16x4: exact fp32 path for the small GEMMs
__device__ __forceinline__ v8f wmma4(v2f a, v2f b, v8f c) {
  return __builtin_amdgcn_wmma_f32_16x16x4_f32(false, a, false, b, (short)0, c, false, false);
}
// f16 WMMA 16x16x32 (fp32 accumulate): bulk pipe for the edge MLP (split-f16)
__device__ __forceinline__ v8f wmma16(v16h a, v16h b, v8f c) {
  return __builtin_amdgcn_wmma_f32_16x16x32_f16(false, a, false, b, (short)0, c, false, false);
}

__device__ __forceinline__ float silu_f(float x) {
  return x * (1.0f / (1.0f + __expf(-x)));
}

// edge id within a group for (src_local=i, tgt_local=j), i!=j, row-major enumeration
__device__ __forceinline__ int edge_global(int srcNode, int tgtNode) {
  int g = srcNode / NPG_;
  int i = srcNode - g * NPG_;
  int j = tgtNode - g * NPG_;
  return g * NPG_ * (NPG_ - 1) + i * (NPG_ - 1) + (j < i ? j : j - 1);
}

// ---------------- generic fp32 WMMA GEMM: C = act(A @ W + bias) ----------------
// A: MxKd row-major, W: KdxN row-major, C: MxN row-major.
// Assumes M%16==0, N%16==0, Kd%4==0 (true for every call below).
__global__ __launch_bounds__(32)
void gemm_kernel(const float* __restrict__ A, const float* __restrict__ W,
                 const float* __restrict__ bias, float* __restrict__ C,
                 int Kd, int N, int act) {
  const int m0   = blockIdx.x * 16;
  const int n0   = blockIdx.y * 16;
  const int lane = threadIdx.x;
  const int half = lane >> 4;
  const int l    = lane & 15;
  const float* Arow = A + (size_t)(m0 + l) * Kd;
  v8f acc = {0.f, 0.f, 0.f, 0.f, 0.f, 0.f, 0.f, 0.f};
  for (int k0 = 0; k0 < Kd; k0 += 4) {
    const int ka = k0 + 2 * half;
    v2f av; av.x = Arow[ka];                       av.y = Arow[ka + 1];
    v2f bv; bv.x = W[(size_t)ka * N + n0 + l];     bv.y = W[(size_t)(ka + 1) * N + n0 + l];
    acc = wmma4(av, bv, acc);
  }
  const int n = n0 + l;
  const float bn = bias ? bias[n] : 0.f;
#pragma unroll
  for (int rr = 0; rr < 8; ++rr) {
    const int m = m0 + rr + half * 8;
    float val = acc[rr] + bn;
    if (act == 1) val = silu_f(val);
    C[(size_t)m * N + n] = val;
  }
}

// ---------------- split fp32 weights into f16 hi/lo pairs (with K padding) -------
__global__ void split_w_kernel(const float* __restrict__ W, half_t* __restrict__ Wh,
                               half_t* __restrict__ Wl, int rows, int prows, int cols) {
  const int i = blockIdx.x * blockDim.x + threadIdx.x;
  if (i >= prows * cols) return;
  const int r = i / cols;
  const float w = (r < rows) ? W[(size_t)r * cols + (i - r * cols)] : 0.f;
  const half_t h = (half_t)w;
  Wh[i] = h;
  Wl[i] = (half_t)(w - (float)h);
}

// ---------------- KNN (47 candidates per node, K=4, stable tie-break) ----------------
__global__ void knn_kernel(const float* __restrict__ p, int* __restrict__ nbr) {
  const int n = blockIdx.x * blockDim.x + threadIdx.x;
  if (n >= NN_) return;
  const int base = (n / NPG_) * NPG_;
  const float px = p[n * 3], py = p[n * 3 + 1], pz = p[n * 3 + 2];
  int chosen[K_];
  for (int k = 0; k < K_; ++k) {
    float best = 3.0e38f; int bestj = base;
    for (int j = 0; j < NPG_; ++j) {
      const int cand = base + j;
      if (cand == n) continue;
      bool used = false;
      for (int q = 0; q < k; ++q) used |= (chosen[q] == cand);
      if (used) continue;
      const float dx = px - p[cand * 3], dy = py - p[cand * 3 + 1], dz = pz - p[cand * 3 + 2];
      const float d2 = dx * dx + dy * dy + dz * dz;
      if (d2 < best) { best = d2; bestj = cand; }   // strict < : lower index wins ties
    }
    chosen[k] = bestj;
    nbr[n * K_ + k] = bestj;
  }
}

// ---------------- triplet features + edge_lin + masked-mean scatter into e1b ----------
__global__ __launch_bounds__(64)
void triplet_kernel(const float* __restrict__ p, const float* __restrict__ e1,
                    const int* __restrict__ nbr, const float* __restrict__ Wlin,
                    const float* __restrict__ blin, float* __restrict__ e1b) {
  const int t   = blockIdx.x;        // knn edge, 0..EK_-1
  const int tid = threadIdx.x;       // 64 threads = ED outputs
  const int ktgt = t / K_;
  const int ksrc = nbr[t];
  __shared__ float f[K_][136];
  __shared__ float sh_mask[K_];
  __shared__ int   sh_eid_kj[K_];
  __shared__ int   sh_eid_main;
  __shared__ float sh_cnt;

  if (tid < K_) {
    const int kk = tid;
    const int nb = nbr[ksrc * K_ + kk];              // idx_k
    sh_mask[kk]  = (ktgt != nb) ? 1.f : 0.f;
    sh_eid_kj[kk] = edge_global(nb, ksrc);
    const float rx = p[ksrc * 3] - p[ktgt * 3];
    const float ry = p[ksrc * 3 + 1] - p[ktgt * 3 + 1];
    const float rz = p[ksrc * 3 + 2] - p[ktgt * 3 + 2];
    const float dm = sqrtf(rx * rx + ry * ry + rz * rz);
    const float im = 1.f / fmaxf(dm, 1e-12f);
    const float qx = p[nb * 3] - p[ksrc * 3];
    const float qy = p[nb * 3 + 1] - p[ksrc * 3 + 1];
    const float qz = p[nb * 3 + 2] - p[ksrc * 3 + 2];
    const float dk = sqrtf(qx * qx + qy * qy + qz * qz);
    const float ik = 1.f / fmaxf(dk, 1e-12f);
    float cosang = -(qx * ik * rx * im + qy * ik * ry * im + qz * ik * rz * im);
    cosang = fminf(fmaxf(cosang, -1.f + 1e-7f), 1.f - 1e-7f);
    f[kk][128] = acosf(cosang);
    f[kk][129] = dm;
    f[kk][130] = dk;
  }
  if (tid == 0) sh_eid_main = edge_global(ksrc, ktgt);
  __syncthreads();
  if (tid == 0) sh_cnt = fmaxf(sh_mask[0] + sh_mask[1] + sh_mask[2] + sh_mask[3], 1.f);
  for (int idx = tid; idx < K_ * 128; idx += 64) {
    const int kk = idx >> 7, c = idx & 127;
    f[kk][c] = (c < ED_) ? e1[(size_t)sh_eid_main * ED_ + c]
                         : e1[(size_t)sh_eid_kj[kk] * ED_ + (c - ED_)];
  }
  __syncthreads();
  float accum = 0.f;
  for (int kk = 0; kk < K_; ++kk) {
    if (sh_mask[kk] > 0.f) {
      float dot = blin[tid];
      for (int c = 0; c < 131; ++c) dot += f[kk][c] * Wlin[c * ED_ + tid];
      accum += dot;
    }
  }
  e1b[(size_t)sh_eid_main * ED_ + tid] += accum / sh_cnt;   // unique pairs: race-free
}

// ---------------- fused edge MLP (split-f16 W1->SiLU->W2) + message scatter ----------
// One workgroup = 16 edges, 8 waves. A tile and h tile live in LDS as f16 hi/lo pairs.
// GEMM = Ah*Bh + Ah*Bl + Al*Bh on v_wmma_f32_16x16x32_f16 (fp32-class accuracy).
__global__ __launch_bounds__(256)
void edge_net_kernel(const float* __restrict__ s, const float* __restrict__ v,
                     const float* __restrict__ p, const float* __restrict__ darr,
                     const float* __restrict__ aarr, const float* __restrict__ rarr,
                     const int* __restrict__ eidx, const float* __restrict__ e1b,
                     const float* __restrict__ sb, const float* __restrict__ vb,
                     const half_t* __restrict__ W1h, const half_t* __restrict__ W1l,
                     const float* __restrict__ b1,
                     const half_t* __restrict__ W2h, const half_t* __restrict__ W2l,
                     const float* __restrict__ b2,
                     float* __restrict__ s2acc, float* __restrict__ v2acc,
                     float* __restrict__ mpacc, float* __restrict__ e2pre) {
  __shared__ half_t Ah[16 * W1KP_];
  __shared__ half_t Al[16 * W1KP_];
  __shared__ half_t Hh[16 * SI_];
  __shared__ half_t Hl[16 * SI_];
  __shared__ int sh_src[16], sh_tgt[16];
  const int eb  = blockIdx.x * 16;
  const int tid = threadIdx.x;
  if (tid < 16) {
    sh_src[tid] = eidx[eb + tid];
    sh_tgt[tid] = eidx[NE_ + eb + tid];
  }
  __syncthreads();
  // build aij_in tile: [s[tgt](256) | s[src](256) | d | a | e1(64) | d_i | d_j | pad]
  for (int idx = tid; idx < 16 * W1KP_; idx += 256) {
    const int m = idx / W1KP_, c = idx % W1KP_;
    const int edge = eb + m, src = sh_src[m], tgt = sh_tgt[m];
    float val;
    if (c < SI_)               val = s[(size_t)tgt * SI_ + c];
    else if (c < 2 * SI_)      val = s[(size_t)src * SI_ + (c - SI_)];
    else if (c == 2 * SI_)     val = darr[edge];
    else if (c == 2 * SI_ + 1) val = aarr[edge];
    else if (c < 2 * SI_ + 2 + ED_) val = e1b[(size_t)edge * ED_ + (c - (2 * SI_ + 2))];
    else if (c == 2 * SI_ + 2 + ED_) {
      const float x = p[tgt * 3], y = p[tgt * 3 + 1], z = p[tgt * 3 + 2];
      val = sqrtf(fmaxf(x * x + y * y + z * z, 1e-6f));
    } else if (c == 2 * SI_ + 3 + ED_) {
      const float x = p[src * 3], y = p[src * 3 + 1], z = p[src * 3 + 2];
      val = sqrtf(fmaxf(x * x + y * y + z * z, 1e-6f));
    } else {
      val = 0.f;                                  // K padding 580..607
    }
    const half_t h = (half_t)val;
    Ah[idx] = h;
    Al[idx] = (half_t)(val - (float)h);
  }
  __syncthreads();

  const int wave = tid >> 5, lane = tid & 31;
  const int half = lane >> 4, l = lane & 15;

  // ---- h = silu(A @ W1 + b1) : M=16, K=608(pad), N=256 (16 n-tiles over 8 waves) ----
  for (int nt = wave; nt < SI_ / 16; nt += 8) {
    const int n0 = nt * 16;
    const int n  = n0 + l;
    v8f acc = {0.f, 0.f, 0.f, 0.f, 0.f, 0.f, 0.f, 0.f};
    for (int kc = 0; kc < W1KP_; kc += 32) {
      const half_t* arh = &Ah[l * W1KP_ + kc];
      const half_t* arl = &Al[l * W1KP_ + kc];
      v16h ah, al, bh, bl;
#pragma unroll
      for (int i = 0; i < 8; ++i) {              // A(16x32): elem i = K 16h+i
        ah[i]     = arh[16 * half + i];
        ah[i + 8] = arh[16 * half + 8 + i];
        al[i]     = arl[16 * half + i];
        al[i + 8] = arl[16 * half + 8 + i];
      }
#pragma unroll
      for (int i = 0; i < 16; ++i) {             // B(32x16): elem i = K 16h+i, lane = N
        const size_t off = (size_t)(kc + 16 * half + i) * SI_ + n;
        bh[i] = W1h[off];
        bl[i] = W1l[off];
      }
      acc = wmma16(ah, bh, acc);
      acc = wmma16(ah, bl, acc);
      acc = wmma16(al, bh, acc);
    }
    const float bn = b1[n];
#pragma unroll
    for (int rr = 0; rr < 8; ++rr) {
      const int m = rr + half * 8;
      const float hv = silu_f(acc[rr] + bn);
      const half_t hh_ = (half_t)hv;
      Hh[m * SI_ + n] = hh_;
      Hl[m * SI_ + n] = (half_t)(hv - (float)hh_);
    }
  }
  __syncthreads();

  // ---- out = h @ W2 + b2 : M=16, K=256, N=513 (33 n-tiles) + fused epilogue ----
  for (int nt = wave; nt < 33; nt += 8) {
    const int n0 = nt * 16;
    const int n = n0 + l;
    const bool nvalid = (n < W2N_);
    v8f acc = {0.f, 0.f, 0.f, 0.f, 0.f, 0.f, 0.f, 0.f};
    for (int kc = 0; kc < SI_; kc += 32) {
      const half_t* arh = &Hh[l * SI_ + kc];
      const half_t* arl = &Hl[l * SI_ + kc];
      v16h ah, al, bh, bl;
#pragma unroll
      for (int i = 0; i < 8; ++i) {
        ah[i]     = arh[16 * half + i];
        ah[i + 8] = arh[16 * half + 8 + i];
        al[i]     = arl[16 * half + i];
        al[i + 8] = arl[16 * half + 8 + i];
      }
#pragma unroll
      for (int i = 0; i < 16; ++i) {
        const size_t off = (size_t)(kc + 16 * half + i) * W2N_ + n;
        bh[i] = nvalid ? W2h[off] : (half_t)0.f;
        bl[i] = nvalid ? W2l[off] : (half_t)0.f;
      }
      acc = wmma16(ah, bh, acc);
      acc = wmma16(ah, bl, acc);
      acc = wmma16(al, bh, acc);
    }
    if (!nvalid) continue;
    const float bn = b2[n];
#pragma unroll
    for (int rr = 0; rr < 8; ++rr) {
      const int m = rr + half * 8;
      const int edge = eb + m;
      const int src = sh_src[m], tgt = sh_tgt[m];
      const float val = acc[rr] + bn;
      if (n < SI_) {                         // aij -> ns = silu(val)*sb[src]
        atomicAdd(&s2acc[(size_t)tgt * SI_ + n], silu_f(val) * sb[(size_t)src * SI_ + n]);
      } else if (n == SI_) {                 // pij -> npv = pij*r
        for (int ax = 0; ax < 3; ++ax)
          atomicAdd(&mpacc[tgt * 3 + ax], val * rarr[edge * 3 + ax]);
      } else if (n < SI_ + 1 + 3 * VI_) {    // vij0/1/2 -> nv
        const int tt = n - (SI_ + 1);
        const int which = tt / VI_, vi = tt % VI_;
        if (which == 0) {
          for (int ax = 0; ax < 3; ++ax)
            atomicAdd(&v2acc[((size_t)tgt * 3 + ax) * VI_ + vi], val * rarr[edge * 3 + ax]);
        } else if (which == 1) {
          for (int ax = 0; ax < 3; ++ax)
            atomicAdd(&v2acc[((size_t)tgt * 3 + ax) * VI_ + vi],
                      val * vb[((size_t)src * 3 + ax) * VI_ + vi]);
        } else {                             // cross(v[tgt], v[src]) per vi
          const float vt0 = v[((size_t)tgt * 3 + 0) * VI_ + vi];
          const float vt1 = v[((size_t)tgt * 3 + 1) * VI_ + vi];
          const float vt2 = v[((size_t)tgt * 3 + 2) * VI_ + vi];
          const float vs0 = v[((size_t)src * 3 + 0) * VI_ + vi];
          const float vs1 = v[((size_t)src * 3 + 1) * VI_ + vi];
          const float vs2 = v[((size_t)src * 3 + 2) * VI_ + vi];
          atomicAdd(&v2acc[((size_t)tgt * 3 + 0) * VI_ + vi], val * (vt1 * vs2 - vt2 * vs1));
          atomicAdd(&v2acc[((size_t)tgt * 3 + 1) * VI_ + vi], val * (vt2 * vs0 - vt0 * vs2));
          atomicAdd(&v2acc[((size_t)tgt * 3 + 2) * VI_ + vi], val * (vt0 * vs1 - vt1 * vs0));
        }
      } else {                               // gij -> e2pre = silu(gij + e1)
        const int g = n - (SI_ + 1 + 3 * VI_);
        const float x = val + e1b[(size_t)edge * ED_ + g];
        e2pre[(size_t)edge * ED_ + g] = silu_f(x);
      }
    }
  }
}

// ---------------- node updates ----------------
__global__ void node_update_kernel(const float* __restrict__ s, const float* __restrict__ s2acc,
                                   float* __restrict__ s2, const float* __restrict__ v,
                                   const float* __restrict__ v2acc, float* __restrict__ v2,
                                   const float* __restrict__ mpacc, float* __restrict__ mp,
                                   float* __restrict__ bsum) {
  const int i = blockIdx.x * blockDim.x + threadIdx.x;
  const float invdeg = 1.f / (float)(NPG_ - 1);   // deg == 47 for every node
  if (i < NN_ * SI_)     s2[i] = s[i] + s2acc[i];
  if (i < NN_ * 3 * VI_) v2[i] = v[i] + v2acc[i] * invdeg;
  if (i < NN_) {
    const float mx = mpacc[i * 3] * invdeg;
    const float my = mpacc[i * 3 + 1] * invdeg;
    const float mz = mpacc[i * 3 + 2] * invdeg;
    mp[i * 3] = mx; mp[i * 3 + 1] = my; mp[i * 3 + 2] = mz;
    atomicAdd(&bsum[i / NPG_], sqrtf(mx * mx + my * my + mz * mz));
  }
}

__global__ void pos_update_kernel(const float* __restrict__ p, const float* __restrict__ mp,
                                  const float* __restrict__ bsum, const float* __restrict__ posw,
                                  float* __restrict__ p2out) {
  const int i = blockIdx.x * blockDim.x + threadIdx.x;
  if (i >= NN_ * 3) return;
  const int n = i / 3;
  const float mean = bsum[n / NPG_] * (1.f / (float)NPG_);
  p2out[i] = p[i] + posw[0] * mp[i] / (mean + 1e-5f);
}

// ---------------- head: hin = [s2 | vnorm] ----------------
__global__ void build_hin_kernel(const float* __restrict__ s2, const float* __restrict__ vv,
                                 float* __restrict__ hin) {
  const int i = blockIdx.x * blockDim.x + threadIdx.x;
  if (i >= NN_ * (SI_ + VI_)) return;
  const int n = i / (SI_ + VI_), c = i % (SI_ + VI_);
  float val;
  if (c < SI_) val = s2[(size_t)n * SI_ + c];
  else {
    const int vi = c - SI_;
    const float a0 = vv[((size_t)n * 3 + 0) * (2 * VI_) + vi];
    const float a1 = vv[((size_t)n * 3 + 1) * (2 * VI_) + vi];
    const float a2 = vv[((size_t)n * 3 + 2) * (2 * VI_) + vi];
    val = sqrtf(fmaxf(a0 * a0 + a1 * a1 + a2 * a2, 1e-6f));
  }
  hin[i] = val;
}

__global__ void final_out_kernel(const float* __restrict__ s2, const float* __restrict__ oo,
                                 const float* __restrict__ v2, const float* __restrict__ vv,
                                 float* __restrict__ s3out, float* __restrict__ v3out) {
  const int i = blockIdx.x * blockDim.x + threadIdx.x;
  if (i < NN_ * SI_) {
    const int n = i / SI_, c = i % SI_;
    s3out[i] = s2[i] + oo[(size_t)n * (VI_ + SI_) + VI_ + c];
  }
  if (i < NN_ * 3 * VI_) {
    const int n = i / (3 * VI_), rem = i % (3 * VI_);
    const int ax = rem / VI_, vi = rem % VI_;
    v3out[i] = v2[i] + oo[(size_t)n * (VI_ + SI_) + vi] *
                           vv[((size_t)n * 3 + ax) * (2 * VI_) + VI_ + vi];
  }
}

// ---------------- workspace layout (float units) ----------------
#define OFF_E1     ((size_t)0)
#define OFF_E1B    (OFF_E1 + (size_t)NE_ * ED_)
#define OFF_NBR    (OFF_E1B + (size_t)NE_ * ED_)
#define OFF_SB     (OFF_NBR + (size_t)NN_ * K_)
#define OFF_VB     (OFF_SB + (size_t)NN_ * SI_)
#define OFF_S2ACC  (OFF_VB + (size_t)NN_ * 3 * VI_)
#define OFF_V2ACC  (OFF_S2ACC + (size_t)NN_ * SI_)
#define OFF_MPACC  (OFF_V2ACC + (size_t)NN_ * 3 * VI_)
#define OFF_BSUM   (OFF_MPACC + (size_t)NN_ * 3)
#define OFF_E2PRE  (OFF_BSUM + (size_t)NB_)
#define OFF_S2     (OFF_E2PRE + (size_t)NE_ * ED_)
#define OFF_V2     (OFF_S2 + (size_t)NN_ * SI_)
#define OFF_MP     (OFF_V2 + (size_t)NN_ * 3 * VI_)
#define OFF_VV     (OFF_MP + (size_t)NN_ * 3)
#define OFF_HIN    (OFF_VV + (size_t)NN_ * 3 * 2 * VI_)
#define OFF_HH     (OFF_HIN + (size_t)NN_ * (SI_ + VI_))
#define OFF_OO     (OFF_HH + (size_t)NN_ * SI_)
#define OFF_WHALF  (OFF_OO + (size_t)NN_ * (VI_ + SI_))   // half_t region starts here

extern "C" void kernel_launch(void* const* d_in, const int* in_sizes, int n_in,
                              void* d_out, int out_size, void* d_ws, size_t ws_size,
                              hipStream_t stream) {
  (void)in_sizes; (void)n_in; (void)out_size; (void)ws_size;
  const float* s   = (const float*)d_in[0];
  const float* v   = (const float*)d_in[1];
  const float* p   = (const float*)d_in[2];
  const float* dd  = (const float*)d_in[3];
  const float* aa  = (const float*)d_in[4];
  const float* rr  = (const float*)d_in[5];
  const float* e   = (const float*)d_in[6];
  const int*   ei  = (const int*)d_in[7];
  // d_in[8] = batch (implied by node index / NPG)
  const float* edge_pre_W = (const float*)d_in[9];
  const float* edge_pre_b = (const float*)d_in[10];
  const float* edge_lin_W = (const float*)d_in[11];
  const float* edge_lin_b = (const float*)d_in[12];
  const float* edge_net_W1 = (const float*)d_in[13];
  const float* edge_net_b1 = (const float*)d_in[14];
  const float* edge_net_W2 = (const float*)d_in[15];
  const float* edge_net_b2 = (const float*)d_in[16];
  const float* edge_post_W = (const float*)d_in[17];
  const float* edge_post_b = (const float*)d_in[18];
  const float* scalar_W = (const float*)d_in[19];
  const float* scalar_b = (const float*)d_in[20];
  const float* vector_W = (const float*)d_in[21];
  const float* posnorm_w = (const float*)d_in[22];
  const float* Wv0   = (const float*)d_in[23];
  const float* Ws_W1 = (const float*)d_in[24];
  const float* Ws_b1 = (const float*)d_in[25];
  const float* Ws_W2 = (const float*)d_in[26];
  const float* Ws_b2 = (const float*)d_in[27];

  float* ws = (float*)d_ws;
  float* e1    = ws + OFF_E1;
  float* e1b   = ws + OFF_E1B;
  int*   nbr   = (int*)(ws + OFF_NBR);
  float* sb    = ws + OFF_SB;
  float* vb    = ws + OFF_VB;
  float* s2acc = ws + OFF_S2ACC;
  float* v2acc = ws + OFF_V2ACC;
  float* mpacc = ws + OFF_MPACC;
  float* bsum  = ws + OFF_BSUM;
  float* e2pre = ws + OFF_E2PRE;
  float* s2    = ws + OFF_S2;
  float* v2    = ws + OFF_V2;
  float* mp    = ws + OFF_MP;
  float* vv    = ws + OFF_VV;
  float* hin   = ws + OFF_HIN;
  float* hh    = ws + OFF_HH;
  float* oo    = ws + OFF_OO;
  half_t* W1h  = (half_t*)(ws + OFF_WHALF);
  half_t* W1l  = W1h + (size_t)W1KP_ * SI_;
  half_t* W2h  = W1l + (size_t)W1KP_ * SI_;
  half_t* W2l  = W2h + (size_t)SI_ * W2N_;

  float* out_s3 = (float*)d_out;                       // N x SI
  float* out_v3 = out_s3 + (size_t)NN_ * SI_;          // N x 3 x VI
  float* out_p2 = out_v3 + (size_t)NN_ * 3 * VI_;      // N x 3
  float* out_e2 = out_p2 + (size_t)NN_ * 3;            // E x ED

  // zero accumulators (must be redone every call)
  hipMemsetAsync(s2acc, 0, (size_t)NN_ * SI_ * 4, stream);
  hipMemsetAsync(v2acc, 0, (size_t)NN_ * 3 * VI_ * 4, stream);
  hipMemsetAsync(mpacc, 0, (size_t)NN_ * 3 * 4, stream);
  hipMemsetAsync(bsum,  0, (size_t)NB_ * 4, stream);

  // split edge-net weights into f16 hi/lo (W1 padded to 608 K-rows)
  split_w_kernel<<<(W1KP_ * SI_ + 255) / 256, 256, 0, stream>>>(edge_net_W1, W1h, W1l,
                                                                W1K_, W1KP_, SI_);
  split_w_kernel<<<(SI_ * W2N_ + 255) / 256, 256, 0, stream>>>(edge_net_W2, W2h, W2l,
                                                               SI_, SI_, W2N_);

  // e1 = e @ edge_pre_W + b   (exact fp32 WMMA)
  gemm_kernel<<<dim3(NE_ / 16, ED_ / 16), 32, 0, stream>>>(e, edge_pre_W, edge_pre_b, e1, ED_, ED_, 0);
  hipMemcpyAsync(e1b, e1, (size_t)NE_ * ED_ * 4, hipMemcpyDeviceToDevice, stream);

  // knn + triplet aggregation into e1b
  knn_kernel<<<(NN_ + 255) / 256, 256, 0, stream>>>(p, nbr);
  triplet_kernel<<<EK_, 64, 0, stream>>>(p, e1, nbr, edge_lin_W, edge_lin_b, e1b);

  // node linears (exact fp32 WMMA)
  gemm_kernel<<<dim3(NN_ / 16, SI_ / 16), 32, 0, stream>>>(s, scalar_W, scalar_b, sb, SI_, SI_, 0);
  gemm_kernel<<<dim3(NN_ * 3 / 16, VI_ / 16), 32, 0, stream>>>(v, vector_W, nullptr, vb, VI_, VI_, 0);

  // fused edge MLP (split-f16 WMMA) + message scatter
  edge_net_kernel<<<NE_ / 16, 256, 0, stream>>>(s, v, p, dd, aa, rr, ei, e1b, sb, vb,
                                                W1h, W1l, edge_net_b1,
                                                W2h, W2l, edge_net_b2,
                                                s2acc, v2acc, mpacc, e2pre);

  // node updates + position update
  node_update_kernel<<<(NN_ * SI_ + 255) / 256, 256, 0, stream>>>(s, s2acc, s2, v, v2acc, v2,
                                                                  mpacc, mp, bsum);
  pos_update_kernel<<<(NN_ * 3 + 255) / 256, 256, 0, stream>>>(p, mp, bsum, posnorm_w, out_p2);

  // e2 = e2pre @ edge_post_W + b  (straight into output)
  gemm_kernel<<<dim3(NE_ / 16, ED_ / 16), 32, 0, stream>>>(e2pre, edge_post_W, edge_post_b,
                                                           out_e2, ED_, ED_, 0);

  // head: vv = v2 @ Wv0 ; hin = [s2|vnorm] ; hh = silu(hin@W1+b1) ; oo = hh@W2+b2
  gemm_kernel<<<dim3(NN_ * 3 / 16, (2 * VI_) / 16), 32, 0, stream>>>(v2, Wv0, nullptr, vv,
                                                                     VI_, 2 * VI_, 0);
  build_hin_kernel<<<(NN_ * (SI_ + VI_) + 255) / 256, 256, 0, stream>>>(s2, vv, hin);
  gemm_kernel<<<dim3(NN_ / 16, SI_ / 16), 32, 0, stream>>>(hin, Ws_W1, Ws_b1, hh,
                                                           SI_ + VI_, SI_, 1);
  gemm_kernel<<<dim3(NN_ / 16, (VI_ + SI_) / 16), 32, 0, stream>>>(hh, Ws_W2, Ws_b2, oo,
                                                                   SI_, VI_ + SI_, 0);
  final_out_kernel<<<(NN_ * SI_ + 255) / 256, 256, 0, stream>>>(s2, oo, v2, vv, out_s3, out_v3);
}
